// FusionModel_15994458210583
// MI455X (gfx1250) — matrix operand order repaired and checked by reading, not compiled, not run on hardware
//
#include <hip/hip_runtime.h>

#define B_    512
#define NPG   62
#define N_    (B_*NPG)     // 31744
#define F_IN  5
#define H_    8
#define C_    3
#define EG    992
#define E_    (B_*EG)      // 507904
#define NB    5
#define DE_DIM 930

typedef _Float16 v8h  __attribute__((ext_vector_type(8)));
typedef _Float16 v16h __attribute__((ext_vector_type(16)));
typedef float    v8f  __attribute__((ext_vector_type(8)));

__device__ __forceinline__ float elu_f(float x){ return x > 0.f ? x : __expf(x) - 1.f; }
// order-preserving float<->int key for integer atomicMax (LDS ds_max_i32)
__device__ __forceinline__ int   fkey(float f){ int b = __float_as_int(f); return b ^ ((b >> 31) & 0x7fffffff); }
__device__ __forceinline__ float fdec(int k){ int b = k ^ ((k >> 31) & 0x7fffffff); return __int_as_float(b); }

// ---------------------------------------------------------------- bn stats
__global__ void k_bn_stats(const float* __restrict__ x,
                           float* __restrict__ mu, float* __restrict__ rstd){
  const int nb = blockIdx.x, t = threadIdx.x;
  __shared__ float red[256];
  float ps[F_IN] = {0.f,0.f,0.f,0.f,0.f};
  float pq[F_IN] = {0.f,0.f,0.f,0.f,0.f};
  const float* xb = x + (size_t)nb * N_ * F_IN;
  for(int r = t; r < N_; r += 256){
    #pragma unroll
    for(int f = 0; f < F_IN; ++f){ float v = xb[r*F_IN + f]; ps[f] += v; pq[f] += v*v; }
  }
  for(int f = 0; f < F_IN; ++f){
    red[t] = ps[f]; __syncthreads();
    for(int s = 128; s > 0; s >>= 1){ if(t < s) red[t] += red[t+s]; __syncthreads(); }
    float sum = red[0]; __syncthreads();
    red[t] = pq[f]; __syncthreads();
    for(int s = 128; s > 0; s >>= 1){ if(t < s) red[t] += red[t+s]; __syncthreads(); }
    float sq = red[0]; __syncthreads();
    if(t == 0){
      float m = sum / (float)N_;
      float v = sq / (float)N_ - m*m;
      mu[nb*F_IN + f]   = m;
      rstd[nb*F_IN + f] = rsqrtf(v + 1e-5f);
    }
  }
}

// ------------------------------------------------- fused per-graph GATConv
__global__ void k_gat(const float* __restrict__ x, const int* __restrict__ ei,
                      const float* __restrict__ mu, const float* __restrict__ rstd,
                      const float* __restrict__ bn_g, const float* __restrict__ bn_b,
                      const float* __restrict__ W,  const float* __restrict__ att_src,
                      const float* __restrict__ att_dst, const float* __restrict__ gbias,
                      float* __restrict__ pooled){
  const int g = blockIdx.x, nb = blockIdx.y, t = threadIdx.x;
  __shared__ float s_xn[NPG][F_IN];
  __shared__ float s_xt[NPG][H_*C_];
  __shared__ float s_als[NPG][H_];
  __shared__ float s_ald[NPG][H_];
  __shared__ int   s_m[NPG][H_];
  __shared__ float s_den[NPG][H_];
  __shared__ float s_acc[NPG][H_*C_];
  __shared__ float s_pool[C_];
  __shared__ float s_W[F_IN*H_*C_];
  __shared__ float s_as[H_*C_], s_ad[H_*C_];

  if(t < F_IN*H_*C_) s_W[t] = W[nb*F_IN*H_*C_ + t];
  if(t < H_*C_){ s_as[t] = att_src[nb*H_*C_ + t]; s_ad[t] = att_dst[nb*H_*C_ + t]; }
  if(t < C_) s_pool[t] = 0.f;

  const int base = g * NPG;
  const float* xb = x + ((size_t)nb * N_ + base) * F_IN;
  for(int i = t; i < NPG*F_IN; i += 256){
    int n = i / F_IN, f = i % F_IN;
    s_xn[n][f] = (xb[n*F_IN + f] - mu[nb*F_IN + f]) * rstd[nb*F_IN + f]
                 * bn_g[nb*F_IN + f] + bn_b[nb*F_IN + f];
  }
  __syncthreads();
  for(int i = t; i < NPG*H_*C_; i += 256){
    int n = i / (H_*C_), o = i % (H_*C_);
    float a = 0.f;
    #pragma unroll
    for(int f = 0; f < F_IN; ++f) a += s_xn[n][f] * s_W[f*H_*C_ + o];
    s_xt[n][o] = a;
    s_acc[n][o] = 0.f;
  }
  __syncthreads();
  for(int i = t; i < NPG*H_; i += 256){
    int n = i / H_, h = i % H_;
    float as = 0.f, ad = 0.f;
    #pragma unroll
    for(int c = 0; c < C_; ++c){
      as += s_xt[n][h*C_ + c] * s_as[h*C_ + c];
      ad += s_xt[n][h*C_ + c] * s_ad[h*C_ + c];
    }
    s_als[n][h] = as; s_ald[n][h] = ad;
    s_m[n][h]  = (int)0x80000000;
    s_den[n][h] = 0.f;
  }
  __syncthreads();

  const int* srcp = ei + (size_t)nb*2*E_ + g*EG;
  const int* dstp = srcp + E_;

  // pass A: segment max (LDS ds_max_i32 with order-preserving keys)
  for(int i = t; i < EG; i += 256){
    int s = srcp[i] - base, d = dstp[i] - base;
    #pragma unroll
    for(int h = 0; h < H_; ++h){
      float e = s_als[s][h] + s_ald[d][h];
      e = e > 0.f ? e : 0.2f*e;
      atomicMax(&s_m[d][h], fkey(e));
    }
  }
  for(int i = t; i < NPG; i += 256){   // self loops
    #pragma unroll
    for(int h = 0; h < H_; ++h){
      float e = s_als[i][h] + s_ald[i][h];
      e = e > 0.f ? e : 0.2f*e;
      atomicMax(&s_m[i][h], fkey(e));
    }
  }
  __syncthreads();

  // pass B: exp + weighted aggregate (LDS ds_add_f32)
  for(int i = t; i < EG; i += 256){
    int s = srcp[i] - base, d = dstp[i] - base;
    #pragma unroll
    for(int h = 0; h < H_; ++h){
      float e = s_als[s][h] + s_ald[d][h];
      e = e > 0.f ? e : 0.2f*e;
      float ex = __expf(e - fdec(s_m[d][h]));
      atomicAdd(&s_den[d][h], ex);
      #pragma unroll
      for(int c = 0; c < C_; ++c) atomicAdd(&s_acc[d][h*C_ + c], ex * s_xt[s][h*C_ + c]);
    }
  }
  for(int i = t; i < NPG; i += 256){   // self loops
    #pragma unroll
    for(int h = 0; h < H_; ++h){
      float e = s_als[i][h] + s_ald[i][h];
      e = e > 0.f ? e : 0.2f*e;
      float ex = __expf(e - fdec(s_m[i][h]));
      atomicAdd(&s_den[i][h], ex);
      #pragma unroll
      for(int c = 0; c < C_; ++c) atomicAdd(&s_acc[i][h*C_ + c], ex * s_xt[i][h*C_ + c]);
    }
  }
  __syncthreads();

  // pass C: head mean + bias + elu + mean pool over graph
  if(t < NPG){
    #pragma unroll
    for(int c = 0; c < C_; ++c){
      float o = 0.f;
      #pragma unroll
      for(int h = 0; h < H_; ++h) o += s_acc[t][h*C_ + c] / s_den[t][h];
      o = o * (1.f/H_) + gbias[nb*C_ + c];
      o = elu_f(o);
      atomicAdd(&s_pool[c], o);
    }
  }
  __syncthreads();
  if(t < C_) pooled[((size_t)nb*B_ + g)*C_ + t] = s_pool[t] * (1.f/NPG);
}

// ----------------------------------------------------------- fg: 15 -> 64
__global__ void k_fg(const float* __restrict__ pooled, const float* __restrict__ fgW,
                     const float* __restrict__ fgb, float* __restrict__ x_out){
  int idx = blockIdx.x*blockDim.x + threadIdx.x;
  if(idx >= B_*64) return;
  int b = idx >> 6, j = idx & 63;
  float a = fgb[j];
  #pragma unroll
  for(int k = 0; k < NB*C_; ++k){
    int band = k / C_, c = k % C_;
    a += pooled[((size_t)band*B_ + b)*C_ + c] * fgW[k*64 + j];
  }
  x_out[idx] = elu_f(a);
}

// ----------------------------------------- LayerNorm + Q/K/V (f16 outputs)
__global__ void k_ln_qkv(const float* __restrict__ de, const float* __restrict__ ln_g,
                         const float* __restrict__ ln_b,
                         const float* __restrict__ qW, const float* __restrict__ qb,
                         const float* __restrict__ kW, const float* __restrict__ kb,
                         const float* __restrict__ vW, const float* __restrict__ vb,
                         _Float16* __restrict__ Qh, _Float16* __restrict__ Kh,
                         _Float16* __restrict__ Vt){
  const int b = blockIdx.x, t = threadIdx.x;
  __shared__ float s_de[DE_DIM];
  __shared__ float red[256];
  const float* row = de + (size_t)b*DE_DIM;
  float ps = 0.f, pq = 0.f;
  for(int k = t; k < DE_DIM; k += 256){ float v = row[k]; s_de[k] = v; ps += v; pq += v*v; }
  red[t] = ps; __syncthreads();
  for(int s = 128; s > 0; s >>= 1){ if(t < s) red[t] += red[t+s]; __syncthreads(); }
  float mean = red[0] / (float)DE_DIM; __syncthreads();
  red[t] = pq; __syncthreads();
  for(int s = 128; s > 0; s >>= 1){ if(t < s) red[t] += red[t+s]; __syncthreads(); }
  float var = red[0] / (float)DE_DIM - mean*mean; __syncthreads();
  float rs = rsqrtf(var + 1e-5f);
  for(int k = t; k < DE_DIM; k += 256) s_de[k] = (s_de[k] - mean)*rs*ln_g[k] + ln_b[k];
  __syncthreads();
  if(t < 192){
    int sec = t >> 6, j = t & 63;
    const float* Wm = sec == 0 ? qW : (sec == 1 ? kW : vW);
    const float* bi = sec == 0 ? qb : (sec == 1 ? kb : vb);
    float a = bi[j];
    for(int k = 0; k < DE_DIM; ++k) a += s_de[k] * Wm[k*64 + j];
    if(sec == 0)      Qh[(size_t)b*64 + j] = (_Float16)(a * 0.125f);  // fold 1/sqrt(64)
    else if(sec == 1) Kh[(size_t)b*64 + j] = (_Float16)a;
    else              Vt[(size_t)j*B_ + b] = (_Float16)a;             // transposed
  }
}

// -------------------------------------------- S = (Q/8) @ K^T  via WMMA f16
__global__ void k_qkt_wmma(const _Float16* __restrict__ Qh, const _Float16* __restrict__ Kh,
                           float* __restrict__ S){
  const int lane = threadIdx.x;
  const int m0 = blockIdx.y*16, n0 = blockIdx.x*16;
  const int lo16 = lane & 15, hi = lane >> 4;
  v8f acc = {};
  #pragma unroll
  for(int kk = 0; kk < 64; kk += 32){
    // A fragment: 16x32 f16, lane m = m0+lo16, K = kb+16*hi .. kb+16*hi+15
    const _Float16* ap = Qh + (size_t)(m0 + lo16)*64 + kk + hi*16;
    v8h alo = *(const v8h*)ap;
    v8h ahi = *(const v8h*)(ap + 8);
    v16h a;
    #pragma unroll
    for(int i = 0; i < 8; ++i){ a[i] = alo[i]; a[8+i] = ahi[i]; }
    // B fragment: 32x16, B[k][n] = K[n][k] -> contiguous 16 halves of row n
    v16h bm = *(const v16h*)(Kh + (size_t)(n0 + lo16)*64 + kk + hi*16);
    acc = __builtin_amdgcn_wmma_f32_16x16x32_f16(false, a, false, bm,
                                                 (short)0, acc, false, false);
  }
  float* outp = S + (size_t)(m0 + hi*8)*B_ + n0 + lo16;
  #pragma unroll
  for(int r = 0; r < 8; ++r) outp[(size_t)r*B_] = acc[r];
}

// ------------------------------------------------------- row softmax -> f16
__global__ void k_softmax(const float* __restrict__ S, _Float16* __restrict__ Ph){
  const int r = blockIdx.x, t = threadIdx.x;
  __shared__ float red[256];
  const float* row = S + (size_t)r*B_;
  float m = -3.4e38f;
  for(int c = t; c < B_; c += 256) m = fmaxf(m, row[c]);
  red[t] = m; __syncthreads();
  for(int s = 128; s > 0; s >>= 1){ if(t < s) red[t] = fmaxf(red[t], red[t+s]); __syncthreads(); }
  float rmax = red[0]; __syncthreads();
  float ps = 0.f;
  for(int c = t; c < B_; c += 256) ps += __expf(row[c] - rmax);
  red[t] = ps; __syncthreads();
  for(int s = 128; s > 0; s >>= 1){ if(t < s) red[t] += red[t+s]; __syncthreads(); }
  float inv = 1.f / red[0]; __syncthreads();
  _Float16* prow = Ph + (size_t)r*B_;
  for(int c = t; c < B_; c += 256) prow[c] = (_Float16)(__expf(row[c] - rmax) * inv);
}

// -------------------------------------------------- x_de = P @ V via WMMA
__global__ void k_pv_wmma(const _Float16* __restrict__ Ph, const _Float16* __restrict__ Vt,
                          float* __restrict__ x_de){
  const int lane = threadIdx.x;
  const int m0 = blockIdx.y*16, n0 = blockIdx.x*16;
  const int lo16 = lane & 15, hi = lane >> 4;
  v8f acc = {};
  for(int kk = 0; kk < B_; kk += 32){
    const _Float16* ap = Ph + (size_t)(m0 + lo16)*B_ + kk + hi*16;
    v8h alo = *(const v8h*)ap;
    v8h ahi = *(const v8h*)(ap + 8);
    v16h a;
    #pragma unroll
    for(int i = 0; i < 8; ++i){ a[i] = alo[i]; a[8+i] = ahi[i]; }
    // B[k][n] = V[k][n] = Vt[n][k] -> contiguous in k
    v16h bm = *(const v16h*)(Vt + (size_t)(n0 + lo16)*B_ + kk + hi*16);
    acc = __builtin_amdgcn_wmma_f32_16x16x32_f16(false, a, false, bm,
                                                 (short)0, acc, false, false);
  }
  float* outp = x_de + (size_t)(m0 + hi*8)*64 + n0 + lo16;
  #pragma unroll
  for(int r = 0; r < 8; ++r) outp[(size_t)r*64] = acc[r];
}

// -------------------------------------------------------------- final head
__global__ void k_final(const float* __restrict__ x_out, const float* __restrict__ x_de,
                        const float* __restrict__ faW, const float* __restrict__ fab,
                        float* __restrict__ out){
  int idx = blockIdx.x*blockDim.x + threadIdx.x;
  if(idx >= B_*C_) return;
  int b = idx / C_, c = idx % C_;
  float a = fab[c];
  const float* xo = x_out + (size_t)b*64;
  const float* xd = x_de  + (size_t)b*64;
  #pragma unroll 4
  for(int k = 0; k < 64; ++k) a += xo[k] * faW[k*C_ + c];
  #pragma unroll 4
  for(int k = 0; k < 64; ++k) a += xd[k] * faW[(64 + k)*C_ + c];
  out[idx] = elu_f(a);
}

extern "C" void kernel_launch(void* const* d_in, const int* in_sizes, int n_in,
                              void* d_out, int out_size, void* d_ws, size_t ws_size,
                              hipStream_t stream){
  (void)in_sizes; (void)n_in; (void)out_size; (void)ws_size;
  const float* x       = (const float*)d_in[0];
  const int*   ei      = (const int*)  d_in[1];
  // d_in[2] = batch (unused: graphs are contiguous blocks of 62 nodes)
  const float* de      = (const float*)d_in[3];
  const float* bn_g    = (const float*)d_in[4];
  const float* bn_b    = (const float*)d_in[5];
  const float* W       = (const float*)d_in[6];
  const float* att_src = (const float*)d_in[7];
  const float* att_dst = (const float*)d_in[8];
  const float* gbias   = (const float*)d_in[9];
  const float* fgW     = (const float*)d_in[10];
  const float* fgb     = (const float*)d_in[11];
  const float* ln_g    = (const float*)d_in[12];
  const float* ln_b    = (const float*)d_in[13];
  const float* qW      = (const float*)d_in[14];
  const float* qb      = (const float*)d_in[15];
  const float* kW      = (const float*)d_in[16];
  const float* kb      = (const float*)d_in[17];
  const float* vW      = (const float*)d_in[18];
  const float* vb      = (const float*)d_in[19];
  const float* faW     = (const float*)d_in[20];
  const float* fab     = (const float*)d_in[21];
  float* out = (float*)d_out;

  char* ws = (char*)d_ws;
  size_t off = 0;
  auto wsalloc = [&](size_t bytes)->char*{
    char* p = ws + off; off = (off + bytes + 255) & ~(size_t)255; return p;
  };
  float*     mu     = (float*)    wsalloc((size_t)NB*F_IN*sizeof(float));
  float*     rstd   = (float*)    wsalloc((size_t)NB*F_IN*sizeof(float));
  float*     pooled = (float*)    wsalloc((size_t)NB*B_*C_*sizeof(float));
  float*     x_out  = (float*)    wsalloc((size_t)B_*64*sizeof(float));
  _Float16*  Qh     = (_Float16*) wsalloc((size_t)B_*64*sizeof(_Float16));
  _Float16*  Kh     = (_Float16*) wsalloc((size_t)B_*64*sizeof(_Float16));
  _Float16*  Vt     = (_Float16*) wsalloc((size_t)64*B_*sizeof(_Float16));
  float*     S      = (float*)    wsalloc((size_t)B_*B_*sizeof(float));
  _Float16*  Ph     = (_Float16*) wsalloc((size_t)B_*B_*sizeof(_Float16));
  float*     x_de   = (float*)    wsalloc((size_t)B_*64*sizeof(float));

  k_bn_stats<<<NB, 256, 0, stream>>>(x, mu, rstd);
  dim3 ggat(B_, NB);
  k_gat<<<ggat, 256, 0, stream>>>(x, ei, mu, rstd, bn_g, bn_b, W, att_src, att_dst,
                                  gbias, pooled);
  k_fg<<<(B_*64 + 255)/256, 256, 0, stream>>>(pooled, fgW, fgb, x_out);
  k_ln_qkv<<<B_, 256, 0, stream>>>(de, ln_g, ln_b, qW, qb, kW, kb, vW, vb, Qh, Kh, Vt);
  dim3 gqk(B_/16, B_/16);
  k_qkt_wmma<<<gqk, 32, 0, stream>>>(Qh, Kh, S);
  k_softmax<<<B_, 256, 0, stream>>>(S, Ph);
  dim3 gpv(64/16, B_/16);
  k_pv_wmma<<<gpv, 32, 0, stream>>>(Ph, Vt, x_de);
  k_final<<<(B_*C_ + 255)/256, 256, 0, stream>>>(x_out, x_de, faW, fab, out);
}
